// P2C_20710332301900
// MI455X (gfx1250) — matrix operand order repaired and checked by reading, not compiled, not run on hardware
//
#include <hip/hip_runtime.h>

typedef __attribute__((ext_vector_type(16))) _Float16 v16h;
typedef __attribute__((ext_vector_type(8)))  _Float16 v8h;
typedef __attribute__((ext_vector_type(8)))  float    v8f;

#define BIGF 3.402823466e+38f

// ---------------------------------------------------------------- CDNA5 async copy
// global_load_async_to_lds_b128: VDST = LDS byte offset VGPR, VADDR = 64-bit
// global address, tracked by ASYNCcnt (ISA 10. async / 15.18.3 op 98).
__device__ __forceinline__ void async_b128(const void* g, void* l) {
  asm volatile("global_load_async_to_lds_b128 %0, %1, off"
               :: "v"((unsigned)(size_t)l), "v"(g)
               : "memory");
}
__device__ __forceinline__ void async_wait0() {
  asm volatile("s_wait_asynccnt 0" ::: "memory");
}

// ---------------------------------------------------------------- WMMA helpers

__device__ __forceinline__ v8f wmma32(v16h a, v16h b, v8f c) {
  return __builtin_amdgcn_wmma_f32_16x16x32_f16(false, a, false, b, (short)0, c,
                                                false, false);
}

// A fragment (16x32 f16): lane&15 = M row; lanes 0-15 hold K 0..7 & 16..23,
// lanes 16-31 hold K 8..15 & 24..31 (ISA 7.12.2).
__device__ __forceinline__ v16h frag_a_lds(const _Float16* tile, int row0,
                                           int lane, int ldt) {
  int m  = row0 + (lane & 15);
  int kb = (lane >> 4) * 8;
  v8h lo = *(const v8h*)(tile + m * ldt + kb);
  v8h hi = *(const v8h*)(tile + m * ldt + 16 + kb);
  v16h f;
#pragma unroll
  for (int e = 0; e < 8; ++e) { f[e] = lo[e]; f[e + 8] = hi[e]; }
  return f;
}

// B fragment (32x16 f16): lane&15 = N col; lanes 0-15 K 0..7 & 16..23, lanes 16-31 K 8..15 & 24..31.
__device__ __forceinline__ v16h frag_b_lds(const _Float16* tile, int col0,
                                           int lane, int ldt) {
  int n  = col0 + (lane & 15);
  int ks = (lane >> 4) * 8;
  v8h lo = *(const v8h*)(tile + n * ldt + ks);
  v8h hi = *(const v8h*)(tile + n * ldt + 16 + ks);
  v16h f;
#pragma unroll
  for (int e = 0; e < 8; ++e) { f[e] = lo[e]; f[e + 8] = hi[e]; }
  return f;
}

__device__ __forceinline__ void atomicMaxF(float* addr, float v) {
  if (v >= 0.f) atomicMax((int*)addr, __float_as_int(v));
  else          atomicMin((unsigned int*)addr, __float_as_uint(v));
}

// async-stage a [rows x 32] f16 tile (row-major, ld=K) into LDS (row stride 40)
__device__ __forceinline__ void issue_tile128(const _Float16* M, int K, int row0,
                                              int k0, _Float16* buf) {
  int s = threadIdx.x;
  {
    int r = s >> 2, c8 = (s & 3) * 8;
    async_b128(M + (size_t)(row0 + r) * K + k0 + c8, buf + r * 40 + c8);
  }
  {
    int s2 = s + 256;
    int r = s2 >> 2, c8 = (s2 & 3) * 8;
    async_b128(M + (size_t)(row0 + r) * K + k0 + c8, buf + r * 40 + c8);
  }
}
__device__ __forceinline__ void issue_tile64(const _Float16* M, int K, int row0,
                                             int k0, _Float16* buf) {
  int s = threadIdx.x;
  int r = s >> 2, c8 = (s & 3) * 8;
  async_b128(M + (size_t)(row0 + r) * K + k0 + c8, buf + r * 40 + c8);
}
__device__ __forceinline__ void issue_tile16(const _Float16* M, int K, int k0,
                                             _Float16* buf) {
  int s = threadIdx.x;
  if (s < 64) {
    int r = s >> 2, c8 = (s & 3) * 8;
    async_b128(M + (size_t)r * K + k0 + c8, buf + r * 40 + c8);
  }
}

// ---------------------------------------------------------------- big GEMM
// C[M,N] = epi(A[M,K](f16) @ W[N,K]^T(f16) + bias)
// grid(N/64, M/128), 256 thr (8 waves: 4x2 of 32x32), async double-buffered LDS
constexpr int EPI_F16 = 0, EPI_RELU = 1, EPI_BNRELU = 2, EPI_MAX = 3;

template <int EPI>
__global__ __launch_bounds__(256) void gemm_big(
    const _Float16* __restrict__ A, const _Float16* __restrict__ W,
    const float* __restrict__ bias, const float* __restrict__ gmm,
    const float* __restrict__ bet, const float* __restrict__ mu,
    const float* __restrict__ var, int K, int N, _Float16* __restrict__ Ch,
    float* __restrict__ Cmax, int npts) {
  __shared__ __align__(16) _Float16 Ah[2][128 * 40];
  __shared__ __align__(16) _Float16 Bh[2][64 * 40];
  __shared__ float smax[4][64];
  const int t = threadIdx.x, lane = t & 31, w = t >> 5, wm = w >> 1, wn = w & 1;
  const int rowBlock = blockIdx.y * 128, colBlock = blockIdx.x * 64;
  v8f acc[2][2];
#pragma unroll
  for (int i = 0; i < 2; ++i)
#pragma unroll
    for (int j = 0; j < 2; ++j)
#pragma unroll
      for (int e = 0; e < 8; ++e) acc[i][j][e] = 0.f;

  issue_tile128(A, K, rowBlock, 0, Ah[0]);
  issue_tile64(W, K, colBlock, 0, Bh[0]);
  async_wait0();
  __syncthreads();

  const int nsteps = K >> 5;
  for (int s = 0; s < nsteps; ++s) {
    const int cur = s & 1, nxt = cur ^ 1;
    if (s + 1 < nsteps) {  // overlap next tile's async DMA with this tile's WMMAs
      issue_tile128(A, K, rowBlock, (s + 1) << 5, Ah[nxt]);
      issue_tile64(W, K, colBlock, (s + 1) << 5, Bh[nxt]);
    }
    v16h a0 = frag_a_lds(Ah[cur], wm * 32, lane, 40);
    v16h a1 = frag_a_lds(Ah[cur], wm * 32 + 16, lane, 40);
    v16h b0 = frag_b_lds(Bh[cur], wn * 32, lane, 40);
    v16h b1 = frag_b_lds(Bh[cur], wn * 32 + 16, lane, 40);
    acc[0][0] = wmma32(a0, b0, acc[0][0]);
    acc[0][1] = wmma32(a0, b1, acc[0][1]);
    acc[1][0] = wmma32(a1, b0, acc[1][0]);
    acc[1][1] = wmma32(a1, b1, acc[1][1]);
    if (s + 1 < nsteps) async_wait0();
    __syncthreads();
  }

  if (EPI == EPI_MAX) {  // fused column max-pool -> atomic max into Cmax[batch][col]
#pragma unroll
    for (int j = 0; j < 2; ++j) {
      float m = -BIGF;
#pragma unroll
      for (int i = 0; i < 2; ++i)
#pragma unroll
        for (int e = 0; e < 8; ++e) m = fmaxf(m, acc[i][j][e]);
      m = fmaxf(m, __shfl_xor(m, 16));
      if (lane < 16) smax[wm][wn * 32 + j * 16 + lane] = m;
    }
    __syncthreads();
    if (t < 64) {
      float v = fmaxf(fmaxf(smax[0][t], smax[1][t]), fmaxf(smax[2][t], smax[3][t]));
      int col = colBlock + t;
      v += bias[col];
      int batch = rowBlock / npts;
      atomicMaxF(&Cmax[(size_t)batch * N + col], v);
    }
  } else {
#pragma unroll
    for (int j = 0; j < 2; ++j)
#pragma unroll
      for (int i = 0; i < 2; ++i)
#pragma unroll
        for (int e = 0; e < 8; ++e) {
          int row = rowBlock + wm * 32 + i * 16 + e + ((lane >> 4) << 3);
          int col = colBlock + wn * 32 + j * 16 + (lane & 15);
          float v = acc[i][j][e] + bias[col];
          if (EPI == EPI_BNRELU) {
            v = (v - mu[col]) * rsqrtf(var[col] + 1e-5f) * gmm[col] + bet[col];
            v = fmaxf(v, 0.f);
          }
          if (EPI == EPI_RELU) v = fmaxf(v, 0.f);
          Ch[(size_t)row * N + col] = (_Float16)v;
        }
  }
}

// ---------------------------------------------------------------- M=16 GEMM (decoder)
// grid(N/128), 8 waves, each wave one 16x16 column tile.  EPI: 0 relu->f16, 1 ->f32
template <int EPI>
__global__ __launch_bounds__(256) void gemm_m16(const _Float16* __restrict__ A,
                                                const _Float16* __restrict__ W,
                                                const float* __restrict__ bias,
                                                int K, int N,
                                                _Float16* __restrict__ Ch,
                                                float* __restrict__ Cf) {
  __shared__ __align__(16) _Float16 Ah[2][16 * 40];
  __shared__ __align__(16) _Float16 Bh[2][128 * 40];
  const int t = threadIdx.x, lane = t & 31, w = t >> 5;
  const int colBlock = blockIdx.x * 128;
  v8f acc;
#pragma unroll
  for (int e = 0; e < 8; ++e) acc[e] = 0.f;

  issue_tile16(A, K, 0, Ah[0]);
  issue_tile128(W, K, colBlock, 0, Bh[0]);
  async_wait0();
  __syncthreads();

  const int nsteps = K >> 5;
  for (int s = 0; s < nsteps; ++s) {
    const int cur = s & 1, nxt = cur ^ 1;
    if (s + 1 < nsteps) {
      issue_tile16(A, K, (s + 1) << 5, Ah[nxt]);
      issue_tile128(W, K, colBlock, (s + 1) << 5, Bh[nxt]);
    }
    v16h a = frag_a_lds(Ah[cur], 0, lane, 40);
    v16h b = frag_b_lds(Bh[cur], w * 16, lane, 40);
    acc = wmma32(a, b, acc);
    if (s + 1 < nsteps) async_wait0();
    __syncthreads();
  }
#pragma unroll
  for (int e = 0; e < 8; ++e) {
    int row = e + ((lane >> 4) << 3);
    int col = colBlock + w * 16 + (lane & 15);
    float v = acc[e] + bias[col];
    if (EPI == 0) { v = fmaxf(v, 0.f); Ch[(size_t)row * N + col] = (_Float16)v; }
    else          { Cf[(size_t)row * N + col] = v; }
  }
}

// ---------------------------------------------------------------- utility kernels

__global__ void k_fill(float* p, int n, float v) {
  int i = blockIdx.x * 256 + threadIdx.x;
  if (i < n) p[i] = v;
}

__global__ void k_f32_to_f16(const float* s, _Float16* d, int n) {
  int i = blockIdx.x * 256 + threadIdx.x;
  if (i < n) d[i] = (_Float16)s[i];
}

// FPS: one block per batch, exact scan semantics (record far, update, argmax)
__global__ __launch_bounds__(256) void k_fps(const float* __restrict__ pts,
                                             int* __restrict__ outIdx) {
  __shared__ float dmin[2048];
  __shared__ float rv[256];
  __shared__ int   ri[256];
  __shared__ int   sFar;
  int b = blockIdx.x, t = threadIdx.x;
  const float* P = pts + (size_t)b * 2048 * 3;
  for (int i = t; i < 2048; i += 256) dmin[i] = 1e10f;
  __syncthreads();
  int far = 0;
  for (int g = 0; g < 128; ++g) {
    if (t == 0) outIdx[b * 128 + g] = far;
    float cx = P[far * 3], cy = P[far * 3 + 1], cz = P[far * 3 + 2];
    float bv = -1.f; int bi = 0;
    for (int i = t; i < 2048; i += 256) {
      float dx = P[i * 3] - cx, dy = P[i * 3 + 1] - cy, dz = P[i * 3 + 2] - cz;
      float nd = fminf(dmin[i], dx * dx + dy * dy + dz * dz);
      dmin[i] = nd;
      if (nd > bv) { bv = nd; bi = i; }
    }
    rv[t] = bv; ri[t] = bi;
    __syncthreads();
    for (int s = 128; s > 0; s >>= 1) {
      if (t < s) {
        if (rv[t + s] > rv[t] || (rv[t + s] == rv[t] && ri[t + s] < ri[t])) {
          rv[t] = rv[t + s]; ri[t] = ri[t + s];
        }
      }
      __syncthreads();
    }
    if (t == 0) sFar = ri[0];
    __syncthreads();
    far = sFar;
  }
}

__global__ void k_gather_centers(const float* pts, const int* idx, float* centers) {
  int b = blockIdx.x, g = threadIdx.x;
  int j = idx[b * 128 + g];
  const float* p = pts + ((size_t)b * 2048 + j) * 3;
  float* o = centers + ((size_t)b * 128 + g) * 3;
  o[0] = p[0]; o[1] = p[1]; o[2] = p[2];
}

// brute-force kNN (db = 2048 pts) per (query, batch) block; emits gathered points
// and/or indices in ascending-distance order (top_k tie -> lowest index).
__global__ __launch_bounds__(256) void k_knn(const float* __restrict__ db,
                                             const float* __restrict__ query,
                                             int qStrideB, int k,
                                             float* __restrict__ outPts,
                                             int* __restrict__ outIdx) {
  __shared__ float dist[2048];
  __shared__ float rv[256];
  __shared__ int   ri[256];
  __shared__ int   win;
  int b = blockIdx.y, q = blockIdx.x, t = threadIdx.x, nq = gridDim.x;
  const float* D = db + (size_t)b * 2048 * 3;
  const float* Q = query + (size_t)b * qStrideB + q * 3;
  float qx = Q[0], qy = Q[1], qz = Q[2];
  for (int i = t; i < 2048; i += 256) {
    float dx = D[i * 3] - qx, dy = D[i * 3 + 1] - qy, dz = D[i * 3 + 2] - qz;
    dist[i] = dx * dx + dy * dy + dz * dz;
  }
  __syncthreads();
  for (int r = 0; r < k; ++r) {
    float bv = BIGF; int bi = 2048;
    for (int i = t; i < 2048; i += 256) {
      float d = dist[i];
      if (d < bv) { bv = d; bi = i; }
    }
    rv[t] = bv; ri[t] = bi;
    __syncthreads();
    for (int s = 128; s > 0; s >>= 1) {
      if (t < s) {
        if (rv[t + s] < rv[t] || (rv[t + s] == rv[t] && ri[t + s] < ri[t])) {
          rv[t] = rv[t + s]; ri[t] = ri[t + s];
        }
      }
      __syncthreads();
    }
    if (t == 0) {
      int j = ri[0];
      win = j;
      dist[j] = BIGF;
      if (outIdx) outIdx[((size_t)b * nq + q) * k + r] = j;
      if (outPts) {
        float* o = outPts + (((size_t)b * nq + q) * k + r) * 3;
        o[0] = D[j * 3]; o[1] = D[j * 3 + 1]; o[2] = D[j * 3 + 2];
      }
    }
    __syncthreads();
    (void)win;
  }
}

// stage 1: h1 = relu(bn(x @ W1^T + b1)) -> f16
__global__ void k_mlp1(const float* __restrict__ x, int xStrideB, int npts,
                       const float* __restrict__ W1, const float* __restrict__ b1,
                       const float* __restrict__ g1, const float* __restrict__ be1,
                       const float* __restrict__ m1, const float* __restrict__ v1,
                       _Float16* __restrict__ out) {
  int b = blockIdx.y, p = blockIdx.x, c = threadIdx.x;
  const float* xp = x + (size_t)b * xStrideB + p * 3;
  float h = xp[0] * W1[c * 3] + xp[1] * W1[c * 3 + 1] + xp[2] * W1[c * 3 + 2] + b1[c];
  h = (h - m1[c]) * rsqrtf(v1[c] + 1e-5f) * g1[c] + be1[c];
  out[((size_t)b * npts + p) * 128 + c] = (_Float16)fmaxf(h, 0.f);
}

__global__ void k_rowmax(const _Float16* __restrict__ h2, int npts,
                         float* __restrict__ gmax) {
  int b = blockIdx.x, c = threadIdx.x;
  float m = -BIGF;
  for (int p = 0; p < npts; ++p)
    m = fmaxf(m, (float)h2[((size_t)b * npts + p) * 256 + c]);
  gmax[b * 256 + c] = m;
}

__global__ void k_concat(const _Float16* __restrict__ h2,
                         const float* __restrict__ gmax, int npts,
                         _Float16* __restrict__ out) {
  size_t r = blockIdx.x;
  int t = threadIdx.x;
  int b = (int)(r / npts);
  out[r * 512 + t]       = (_Float16)gmax[b * 256 + t];
  out[r * 512 + 256 + t] = h2[r * 256 + t];
}

// chamfer one direction: sum over a of sqrt(min_b d^2); per-block partial
__global__ __launch_bounds__(256) void k_chamfer(const float* __restrict__ A,
                                                 int na, int aStrideB,
                                                 const float* __restrict__ Bp,
                                                 int nb, int bStrideB,
                                                 float* __restrict__ part) {
  __shared__ float bx[1024], by[1024], bz[1024];
  __shared__ float sm[256];
  int b = blockIdx.y, t = threadIdx.x;
  int ai = blockIdx.x * 256 + t;
  bool act = ai < na;
  const float* Ab = A + (size_t)b * aStrideB;
  float ax = 0, ay = 0, az = 0;
  if (act) { ax = Ab[ai * 3]; ay = Ab[ai * 3 + 1]; az = Ab[ai * 3 + 2]; }
  const float* Bb = Bp + (size_t)b * bStrideB;
  float best = BIGF;
  for (int c0 = 0; c0 < nb; c0 += 1024) {
    int cnt = (nb - c0 < 1024) ? (nb - c0) : 1024;
    for (int i = t; i < cnt; i += 256) {
      const float* q = Bb + (size_t)(c0 + i) * 3;
      bx[i] = q[0]; by[i] = q[1]; bz[i] = q[2];
    }
    __syncthreads();
    if (act)
      for (int i = 0; i < cnt; ++i) {
        float dx = ax - bx[i], dy = ay - by[i], dz = az - bz[i];
        best = fminf(best, dx * dx + dy * dy + dz * dz);
      }
    __syncthreads();
  }
  sm[t] = act ? sqrtf(fmaxf(best, 1e-12f)) : 0.f;
  __syncthreads();
  for (int s = 128; s > 0; s >>= 1) {
    if (t < s) sm[t] += sm[t + s];
    __syncthreads();
  }
  if (t == 0) part[blockIdx.y * gridDim.x + blockIdx.x] = sm[0];
}

__global__ void k_smoothl1(const float* a, const float* b, int n, float* part) {
  __shared__ float sm[256];
  int t = threadIdx.x, i = blockIdx.x * 256 + t;
  float v = 0.f;
  if (i < n) {
    float d = a[i] - b[i], ad = fabsf(d);
    v = (ad < 1.f) ? 0.5f * d * d : ad - 0.5f;
  }
  sm[t] = v;
  __syncthreads();
  for (int s = 128; s > 0; s >>= 1) {
    if (t < s) sm[t] += sm[t + s];
    __syncthreads();
  }
  if (t == 0) part[blockIdx.x] = sm[0];
}

// normals: 3x3 covariance of 32-NN, analytic smallest eigenvector, sign-orient
__global__ __launch_bounds__(256) void k_normals(const float* __restrict__ pred,
                                                 const int* __restrict__ idx,
                                                 float* __restrict__ nrm) {
  int b = blockIdx.y, p = blockIdx.x * 256 + threadIdx.x;
  const float* P = pred + (size_t)b * 6144;
  const int* id = idx + ((size_t)b * 2048 + p) * 32;
  float sx = 0, sy = 0, sz = 0, sxx = 0, sxy = 0, sxz = 0, syy = 0, syz = 0, szz = 0;
  for (int k = 0; k < 32; ++k) {
    int j = id[k];
    float x = P[j * 3], y = P[j * 3 + 1], z = P[j * 3 + 2];
    sx += x; sy += y; sz += z;
    sxx += x * x; sxy += x * y; sxz += x * z;
    syy += y * y; syz += y * z; szz += z * z;
  }
  const float iK = 1.f / 32.f;
  float mx = sx * iK, my = sy * iK, mz = sz * iK;
  float a00 = sxx * iK - mx * mx, a01 = sxy * iK - mx * my, a02 = sxz * iK - mx * mz;
  float a11 = syy * iK - my * my, a12 = syz * iK - my * mz, a22 = szz * iK - mz * mz;
  float nx = 1.f, ny = 0.f, nz = 0.f;
  float p1 = a01 * a01 + a02 * a02 + a12 * a12;
  float q = (a00 + a11 + a22) / 3.f;
  float p2 = (a00 - q) * (a00 - q) + (a11 - q) * (a11 - q) + (a22 - q) * (a22 - q) + 2.f * p1;
  float pp = sqrtf(fmaxf(p2 / 6.f, 0.f));
  if (pp > 1e-20f) {
    float ip = 1.f / pp;
    float b00 = (a00 - q) * ip, b11 = (a11 - q) * ip, b22 = (a22 - q) * ip;
    float b01 = a01 * ip, b02 = a02 * ip, b12 = a12 * ip;
    float detB = b00 * (b11 * b22 - b12 * b12) - b01 * (b01 * b22 - b12 * b02) +
                 b02 * (b01 * b12 - b11 * b02);
    float r = fminf(1.f, fmaxf(-1.f, 0.5f * detB));
    float lam = q + 2.f * pp * cosf(acosf(r) * (1.f / 3.f) + 2.0943951023931953f);
    float r0x = a00 - lam, r0y = a01, r0z = a02;
    float r1x = a01, r1y = a11 - lam, r1z = a12;
    float r2x = a02, r2y = a12, r2z = a22 - lam;
    float c0x = r0y * r1z - r0z * r1y, c0y = r0z * r1x - r0x * r1z, c0z = r0x * r1y - r0y * r1x;
    float c1x = r0y * r2z - r0z * r2y, c1y = r0z * r2x - r0x * r2z, c1z = r0x * r2y - r0y * r2x;
    float c2x = r1y * r2z - r1z * r2y, c2y = r1z * r2x - r1x * r2z, c2z = r1x * r2y - r1y * r2x;
    float n0 = c0x * c0x + c0y * c0y + c0z * c0z;
    float n1 = c1x * c1x + c1y * c1y + c1z * c1z;
    float n2 = c2x * c2x + c2y * c2y + c2z * c2z;
    float bxv = c0x, byv = c0y, bzv = c0z, bn = n0;
    if (n1 > bn) { bxv = c1x; byv = c1y; bzv = c1z; bn = n1; }
    if (n2 > bn) { bxv = c2x; byv = c2y; bzv = c2z; bn = n2; }
    if (bn > 1e-30f) { float inv = rsqrtf(bn); nx = bxv * inv; ny = byv * inv; nz = bzv * inv; }
  }
  float px = P[p * 3], py = P[p * 3 + 1], pz = P[p * 3 + 2];
  float proj = (mx - px) * nx + (my - py) * ny + (mz - pz) * nz;
  float s = (proj >= 0.f) ? 1.f : -1.f;
  float* o = nrm + ((size_t)b * 2048 + p) * 3;
  o[0] = nx * s; o[1] = ny * s; o[2] = nz * s;
}

__global__ __launch_bounds__(256) void k_manifold(const float* __restrict__ nrm,
                                                  const int* __restrict__ idx,
                                                  float* __restrict__ part) {
  __shared__ float sm[256];
  int b = blockIdx.y, t = threadIdx.x, p = blockIdx.x * 256 + t;
  const float* Nn = nrm + (size_t)b * 6144;
  const int* id = idx + ((size_t)b * 2048 + p) * 32;
  int j0 = id[0];
  float ax = Nn[j0 * 3], ay = Nn[j0 * 3 + 1], az = Nn[j0 * 3 + 2];
  float an = fmaxf(sqrtf(ax * ax + ay * ay + az * az), 1e-6f);
  float x[8], mean = 0.f;
#pragma unroll
  for (int k = 0; k < 8; ++k) {
    int j = id[k];
    float bx = Nn[j * 3], by = Nn[j * 3 + 1], bz = Nn[j * 3 + 2];
    float bn = fmaxf(sqrtf(bx * bx + by * by + bz * bz), 1e-6f);
    float c = (ax * bx + ay * by + az * bz) / (an * bn);
    x[k] = 1.f - c;
    mean += x[k];
  }
  mean *= 0.125f;
  float var = 0.f;
#pragma unroll
  for (int k = 0; k < 8; ++k) { float d = x[k] - mean; var += d * d; }
  sm[t] = sqrtf(var / 7.f);
  __syncthreads();
  for (int s = 128; s > 0; s >>= 1) {
    if (t < s) sm[t] += sm[t + s];
    __syncthreads();
  }
  if (t == 0) part[blockIdx.y * gridDim.x + blockIdx.x] = sm[0];
}

__global__ void k_reduce_sum(const float* part, int n, float* out) {
  __shared__ float sm[256];
  int t = threadIdx.x;
  float s = 0.f;
  for (int i = t; i < n; i += 256) s += part[i];
  sm[t] = s;
  __syncthreads();
  for (int st = 128; st > 0; st >>= 1) {
    if (t < st) sm[t] += sm[t + st];
    __syncthreads();
  }
  if (t == 0) out[0] = sm[0];
}

__global__ void k_finalize(const float* accum, float* out) {
  float l_recon = 0.5f * (accum[0] / (16.f * 2048.f) + accum[1] / (16.f * 2432.f));
  float l_match = 0.5f * (accum[2] / (16.f * 1024.f) + accum[3] / (16.f * 1216.f));
  float l_latent = accum[4] / (16.f * 1024.f);
  float l_man = 0.1f * accum[5] / (16.f * 2048.f);
  out[0] = l_recon + l_match + l_latent + l_man;
  out[1] = l_recon;
  out[2] = l_match;
  out[3] = l_latent;
  out[4] = l_man;
}

// ---------------------------------------------------------------- launch

extern "C" void kernel_launch(void* const* d_in, const int* in_sizes, int n_in,
                              void* d_out, int out_size, void* d_ws, size_t ws_size,
                              hipStream_t stream) {
  (void)in_sizes; (void)n_in; (void)out_size; (void)ws_size;
  const float* pts = (const float*)d_in[0];
  const float* W1 = (const float*)d_in[1];
  const float* b1 = (const float*)d_in[2];
  const float* g1 = (const float*)d_in[3];
  const float* be1 = (const float*)d_in[4];
  const float* m1 = (const float*)d_in[5];
  const float* v1 = (const float*)d_in[6];
  const float* W2 = (const float*)d_in[7];
  const float* b2 = (const float*)d_in[8];
  const float* W3 = (const float*)d_in[9];
  const float* b3 = (const float*)d_in[10];
  const float* g2 = (const float*)d_in[11];
  const float* be2 = (const float*)d_in[12];
  const float* m2 = (const float*)d_in[13];
  const float* v2 = (const float*)d_in[14];
  const float* W4 = (const float*)d_in[15];
  const float* b4 = (const float*)d_in[16];
  const float* D1W = (const float*)d_in[17];
  const float* D1b = (const float*)d_in[18];
  const float* D2W = (const float*)d_in[19];
  const float* D2b = (const float*)d_in[20];
  const float* D3W = (const float*)d_in[21];
  const float* D3b = (const float*)d_in[22];
  const float* D4W = (const float*)d_in[23];
  const float* D4b = (const float*)d_in[24];

  char* wsp = (char*)d_ws;
  size_t off = 0;
  auto alloc = [&](size_t bytes) -> char* {
    char* p = wsp + off;
    off = (off + bytes + 255) & ~(size_t)255;
    return p;
  };
  float*    accum   = (float*)alloc(6 * 4);
  float*    part    = (float*)alloc(6 * 256 * 4);
  int*      fpsIdx  = (int*)alloc(16 * 128 * 4);
  float*    centers = (float*)alloc(16 * 128 * 3 * 4);
  float*    grouped = (float*)alloc((size_t)16 * 96 * 32 * 3 * 4);
  int*      knn32   = (int*)alloc((size_t)16 * 2048 * 32 * 4);
  float*    nrm     = (float*)alloc((size_t)16 * 2048 * 3 * 4);
  _Float16* h1      = (_Float16*)alloc((size_t)32768 * 128 * 2);
  _Float16* h2      = (_Float16*)alloc((size_t)32768 * 256 * 2);
  float*    gmax    = (float*)alloc(16 * 256 * 4);
  _Float16* h3in    = (_Float16*)alloc((size_t)32768 * 512 * 2);
  _Float16* h3out   = (_Float16*)alloc((size_t)32768 * 512 * 2);
  float*    feat    = (float*)alloc(16 * 1024 * 4);
  float*    featr   = (float*)alloc(16 * 1024 * 4);
  _Float16* feath   = (_Float16*)alloc(16 * 1024 * 2);
  _Float16* dbuf1   = (_Float16*)alloc(16 * 2048 * 2);
  _Float16* dbuf2   = (_Float16*)alloc(16 * 2048 * 2);
  float*    pred    = (float*)alloc(16 * 6144 * 4);
  float*    Y1      = (float*)alloc((size_t)16 * 2432 * 3 * 4);
  float*    Y2      = (float*)alloc((size_t)16 * 1216 * 3 * 4);
  float*    enc2in  = (float*)alloc((size_t)16 * 1024 * 3 * 4);
  // f16 weight copies (async-copyable, all resident in 192MB L2)
  _Float16* W2h  = (_Float16*)alloc((size_t)256 * 128 * 2);
  _Float16* W3h  = (_Float16*)alloc((size_t)512 * 512 * 2);
  _Float16* W4h  = (_Float16*)alloc((size_t)1024 * 512 * 2);
  _Float16* D1Wh = (_Float16*)alloc((size_t)2048 * 1024 * 2);
  _Float16* D2Wh = (_Float16*)alloc((size_t)2048 * 2048 * 2);
  _Float16* D3Wh = (_Float16*)alloc((size_t)2048 * 2048 * 2);
  _Float16* D4Wh = (_Float16*)alloc((size_t)6144 * 2048 * 2);

  k_f32_to_f16<<<128, 256, 0, stream>>>(W2, W2h, 32768);
  k_f32_to_f16<<<1024, 256, 0, stream>>>(W3, W3h, 262144);
  k_f32_to_f16<<<2048, 256, 0, stream>>>(W4, W4h, 524288);
  k_f32_to_f16<<<8192, 256, 0, stream>>>(D1W, D1Wh, 2097152);
  k_f32_to_f16<<<16384, 256, 0, stream>>>(D2W, D2Wh, 4194304);
  k_f32_to_f16<<<16384, 256, 0, stream>>>(D3W, D3Wh, 4194304);
  k_f32_to_f16<<<49152, 256, 0, stream>>>(D4W, D4Wh, 12582912);

  k_fill<<<64, 256, 0, stream>>>(feat, 16384, -BIGF);
  k_fill<<<64, 256, 0, stream>>>(featr, 16384, -BIGF);

  // grouping
  k_fps<<<16, 256, 0, stream>>>(pts, fpsIdx);
  k_gather_centers<<<16, 128, 0, stream>>>(pts, fpsIdx, centers);
  k_knn<<<dim3(96, 16), 256, 0, stream>>>(pts, centers, 384, 32, grouped, nullptr);

  // encoder pass 1 (rebuild0: groups 0..63 -> 2048 pts/batch, M = 32768)
  k_mlp1<<<dim3(2048, 16), 128, 0, stream>>>(grouped, 9216, 2048, W1, b1, g1, be1, m1, v1, h1);
  gemm_big<EPI_F16><<<dim3(4, 256), 256, 0, stream>>>(h1, W2h, b2, nullptr, nullptr, nullptr, nullptr, 128, 256, h2, nullptr, 2048);
  k_rowmax<<<16, 256, 0, stream>>>(h2, 2048, gmax);
  k_concat<<<32768, 256, 0, stream>>>(h2, gmax, 2048, h3in);
  gemm_big<EPI_BNRELU><<<dim3(8, 256), 256, 0, stream>>>(h3in, W3h, b3, g2, be2, m2, v2, 512, 512, h3out, nullptr, 2048);
  gemm_big<EPI_MAX><<<dim3(16, 256), 256, 0, stream>>>(h3out, W4h, b4, nullptr, nullptr, nullptr, nullptr, 512, 1024, nullptr, feat, 2048);

  // decoder (M = 16)
  k_f32_to_f16<<<64, 256, 0, stream>>>(feat, feath, 16384);
  gemm_m16<0><<<16, 256, 0, stream>>>(feath, D1Wh, D1b, 1024, 2048, dbuf1, nullptr);
  gemm_m16<0><<<16, 256, 0, stream>>>(dbuf1, D2Wh, D2b, 2048, 2048, dbuf2, nullptr);
  gemm_m16<0><<<16, 256, 0, stream>>>(dbuf2, D3Wh, D3b, 2048, 2048, dbuf1, nullptr);
  gemm_m16<1><<<48, 256, 0, stream>>>(dbuf1, D4Wh, D4b, 2048, 6144, nullptr, pred);

  // chamfer recon (rebuild0 vs gather(pred, knn(cg0, pred, 38)))
  k_knn<<<dim3(64, 16), 256, 0, stream>>>(pred, centers, 384, 38, Y1, nullptr);
  k_chamfer<<<dim3(8, 16), 256, 0, stream>>>(grouped, 2048, 9216, Y1, 2432, 7296, part + 0 * 256);
  k_chamfer<<<dim3(10, 16), 256, 0, stream>>>(Y1, 2432, 7296, grouped, 2048, 9216, part + 1 * 256);

  // chamfer match (rebuild1 vs gather(pred, knn(cg1, pred, 38)))
  k_knn<<<dim3(32, 16), 256, 0, stream>>>(pred, centers + 192, 384, 38, Y2, nullptr);
  k_chamfer<<<dim3(4, 16), 256, 0, stream>>>(grouped + 6144, 1024, 9216, Y2, 1216, 3648, part + 2 * 256);
  k_chamfer<<<dim3(5, 16), 256, 0, stream>>>(Y2, 1216, 3648, grouped + 6144, 1024, 9216, part + 3 * 256);

  // encoder pass 2 (gather(pred, knn(cg2, pred, 32)) -> 1024 pts/batch, M = 16384)
  k_knn<<<dim3(32, 16), 256, 0, stream>>>(pred, centers + 288, 384, 32, enc2in, nullptr);
  k_mlp1<<<dim3(1024, 16), 128, 0, stream>>>(enc2in, 3072, 1024, W1, b1, g1, be1, m1, v1, h1);
  gemm_big<EPI_F16><<<dim3(4, 128), 256, 0, stream>>>(h1, W2h, b2, nullptr, nullptr, nullptr, nullptr, 128, 256, h2, nullptr, 1024);
  k_rowmax<<<16, 256, 0, stream>>>(h2, 1024, gmax);
  k_concat<<<16384, 256, 0, stream>>>(h2, gmax, 1024, h3in);
  gemm_big<EPI_BNRELU><<<dim3(8, 128), 256, 0, stream>>>(h3in, W3h, b3, g2, be2, m2, v2, 512, 512, h3out, nullptr, 1024);
  gemm_big<EPI_MAX><<<dim3(16, 128), 256, 0, stream>>>(h3out, W4h, b4, nullptr, nullptr, nullptr, nullptr, 512, 1024, nullptr, featr, 1024);

  // latent smooth-L1
  k_smoothl1<<<64, 256, 0, stream>>>(feat, featr, 16384, part + 4 * 256);

  // normals + manifold
  k_knn<<<dim3(2048, 16), 256, 0, stream>>>(pred, pred, 6144, 32, nullptr, knn32);
  k_normals<<<dim3(8, 16), 256, 0, stream>>>(pred, knn32, nrm);
  k_manifold<<<dim3(8, 16), 256, 0, stream>>>(nrm, knn32, part + 5 * 256);

  // deterministic final reductions
  k_reduce_sum<<<1, 256, 0, stream>>>(part + 0 * 256, 128, accum + 0);
  k_reduce_sum<<<1, 256, 0, stream>>>(part + 1 * 256, 160, accum + 1);
  k_reduce_sum<<<1, 256, 0, stream>>>(part + 2 * 256, 64,  accum + 2);
  k_reduce_sum<<<1, 256, 0, stream>>>(part + 3 * 256, 80,  accum + 3);
  k_reduce_sum<<<1, 256, 0, stream>>>(part + 4 * 256, 64,  accum + 4);
  k_reduce_sum<<<1, 256, 0, stream>>>(part + 5 * 256, 128, accum + 5);

  k_finalize<<<1, 1, 0, stream>>>(accum, (float*)d_out);
}
